// WaveletWindowAttention_46420006535774
// MI455X (gfx1250) — matrix operand order repaired and checked by reading, not compiled, not run on hardware
//
#include <hip/hip_runtime.h>

typedef _Float16 f16;
typedef __attribute__((ext_vector_type(16))) _Float16 v16h;
typedef __attribute__((ext_vector_type(8)))  _Float16 v8h;
typedef __attribute__((ext_vector_type(8)))  float    v8f;
typedef __attribute__((ext_vector_type(4)))  int      v4i;

typedef __attribute__((address_space(1))) v4i* gv4ip;  // global int4*
typedef __attribute__((address_space(3))) v4i* lv4ip;  // LDS int4*

#if defined(__HIP_DEVICE_COMPILE__) && defined(__has_builtin)
#  if __has_builtin(__builtin_amdgcn_global_load_async_to_lds_b128) && \
      __has_builtin(__builtin_amdgcn_s_wait_asynccnt)
#    define USE_ASYNC 1
#  endif
#endif
#ifndef USE_ASYNC
#  define USE_ASYNC 0
#endif

#define DIMC 256
#define RESF 256
#define H2   128
#define NWIN_PER_B 256
#define NWIN 512
#define HEADS 8
#define HD 32
#define NPIX 64
#define SCALE 0.17677669529663687f   /* 32^-0.5 */

/* workspace float offsets (window layout tensors: [win][c][64], 512*256*64) */
#define OFF_LL    0ull
#define OFF_LH    8388608ull
#define OFF_HL    16777216ull
#define OFF_HH    25165824ull
#define OFF_LLOUT 33554432ull
#define OFF_WP16_BYTES 167772160ull   /* f16[65536] after the 5 float tensors */

// ---------------------------------------------------------------- WT forward
__global__ __launch_bounds__(256) void wt_fwd(
    const float* __restrict__ x, const float* __restrict__ wtf,
    float* __restrict__ ll, float* __restrict__ lh,
    float* __restrict__ hl, float* __restrict__ hh) {
  int pix = blockIdx.x * 256 + threadIdx.x;       // 0..16383
  int c = blockIdx.y, b = blockIdx.z;
  int i = pix >> 7, j = pix & 127;
  const float* xp = x + (((size_t)(b * DIMC + c)) * RESF + 2 * i) * RESF + 2 * j;
  float x00 = xp[0], x01 = xp[1], x10 = xp[RESF], x11 = xp[RESF + 1];
  const float* wf = wtf + (size_t)c * 16;         // 4 filters x 2x2 per channel
  float acc[4];
#pragma unroll
  for (int k = 0; k < 4; ++k)
    acc[k] = x00 * wf[k * 4 + 0] + x01 * wf[k * 4 + 1] +
             x10 * wf[k * 4 + 2] + x11 * wf[k * 4 + 3];
  int win = b * NWIN_PER_B + (i >> 3) * 16 + (j >> 3);
  size_t dst = (size_t)win * 16384 + (size_t)c * 64 + ((i & 7) * 8 + (j & 7));
  ll[dst] = acc[0]; lh[dst] = acc[1]; hl[dst] = acc[2]; hh[dst] = acc[3];
}

// ----------------------------------------------------- proj weight f32->f16
__global__ __launch_bounds__(256) void cvt_w16(const float* __restrict__ w,
                                               f16* __restrict__ o, int n) {
  int t = blockIdx.x * 256 + threadIdx.x;
  if (t < n) o[t] = (f16)w[t];
}

// ------------------------------------------------- fused window attn + proj
__global__ __launch_bounds__(256) void win_attn(
    const float* __restrict__ llw, const float* __restrict__ lhw,
    const float* __restrict__ hlw,
    const float* __restrict__ dww, const float* __restrict__ dwb,
    const f16* __restrict__ wp16, const float* __restrict__ pbias,
    const float* __restrict__ abias, const int* __restrict__ bidx,
    int n_off, float* __restrict__ llout) {
  __shared__ f16 s_xh[NPIX * DIMC];                 // 32KB, proj input [n][c]
  __shared__ float s_lli[HD * NPIX];                // 8KB, running ll_i [d][m]
  __shared__ __align__(16) float s_R[4096];         // 16KB phase-overlapped
  float* s_lh    = s_R;                             // f32[2048] conv input [d][p]
  f16*   s_q     = (f16*)(s_R + 2048);              // f16[2048] q as [p][d]
  f16*   s_kT    = (f16*)(s_R + 3072);              // f16[2048] k as [p][d]
  f16*   s_attn  = (f16*)s_R;                       // f16[4096] attn [n][m]
  f16*   s_attnh = (f16*)(s_R + 2048);              // f16[4096] softmax [n][m]

  int t = threadIdx.x;
  int lane = t & 31, wv = t >> 5;
  int l15 = lane & 15, half = lane >> 4;
  int win = blockIdx.x;
  size_t wbase = (size_t)win * 16384;

  // head-0 ll slice
#pragma unroll
  for (int r = 0; r < 8; ++r) s_lli[t + 256 * r] = llw[wbase + t + 256 * r];

#if USE_ASYNC
  // stream head-0 lhw slice into s_lh via ASYNCcnt path (b128 per lane)
#pragma unroll
  for (int r = 0; r < 2; ++r) {
    int e = (r * 256 + t) * 4;
    __builtin_amdgcn_global_load_async_to_lds_b128(
        (gv4ip)(lhw + wbase + e), (lv4ip)(s_lh + e), 0, 0);
  }
#endif

  for (int head = 0; head < HEADS; ++head) {
    size_t sbase = wbase + (size_t)head * 2048;
#if USE_ASYNC
    __builtin_amdgcn_s_wait_asynccnt(0);   // this wave's async copies landed
#endif
    __syncthreads();                       // everyone's copies landed / prev iter done
    if (head < HEADS - 1)  // CDNA5 global_prefetch_b8 for next head's k slice
      __builtin_prefetch(&hlw[sbase + 2048 + t * 8], 0, 0);
    if (head > 0) {
#pragma unroll
      for (int r = 0; r < 8; ++r) s_lli[t + 256 * r] += llw[sbase + t + 256 * r];
    }
#pragma unroll
    for (int r = 0; r < 8; ++r) {
      int e = t + 256 * r;
#if !USE_ASYNC
      s_lh[e] = lhw[sbase + e];
#endif
      int d = e >> 6, p = e & 63;
      s_kT[p * 32 + d] = (f16)hlw[sbase + e];       // transpose to [p][d]
    }
    __syncthreads();

    // depthwise 5x5 conv (pad 2) -> q in [p][d] f16; thread = (channel, row)
    {
      int d = t >> 3, py = t & 7;
      float wreg[25];
#pragma unroll
      for (int u = 0; u < 25; ++u) wreg[u] = dww[(size_t)head * 800 + d * 25 + u];
      float bb = dwb[head * HD + d];
      const float* src = s_lh + d * 64;
#pragma unroll
      for (int px = 0; px < 8; ++px) {
        float acc = bb;
#pragma unroll
        for (int dy = 0; dy < 5; ++dy) {
          int yy = py + dy - 2;
          if (yy < 0 || yy > 7) continue;
#pragma unroll
          for (int dx = 0; dx < 5; ++dx) {
            int xx = px + dx - 2;
            if (xx < 0 || xx > 7) continue;
            acc += src[yy * 8 + xx] * wreg[dy * 5 + dx];
          }
        }
        s_q[(py * 8 + px) * 32 + d] = (f16)acc;
      }
    }
    __syncthreads();

    // attn = Q(64x32) x K(32x64): 2 tiles/wave, single K=32 WMMA
    v8f catt[2];
    int tM[2], tN[2];
#pragma unroll
    for (int tt = 0; tt < 2; ++tt) {
      int tile = wv * 2 + tt;
      int mi = tile >> 2, ni = tile & 3;
      tM[tt] = mi * 16; tN[tt] = ni * 16;
      int m = mi * 16 + l15, n = ni * 16 + l15;
      v8h alo = *(const v8h*)&s_q[m * 32 + half * 8];
      v8h ahi = *(const v8h*)&s_q[m * 32 + 16 + half * 8];
      v8h blo = *(const v8h*)&s_kT[n * 32 + half * 8];
      v8h bhi = *(const v8h*)&s_kT[n * 32 + 16 + half * 8];
      v16h a, b;
#pragma unroll
      for (int u = 0; u < 8; ++u) { a[u] = alo[u]; a[8 + u] = ahi[u];
                                    b[u] = blo[u]; b[8 + u] = bhi[u]; }
      v8f cz = {};
      catt[tt] = __builtin_amdgcn_wmma_f32_16x16x32_f16(false, a, false, b,
                                                        (short)0, cz, false, false);
    }
    __syncthreads();  // all waves done reading s_q/s_kT (aliased by s_attnh)
#pragma unroll
    for (int tt = 0; tt < 2; ++tt) {
#pragma unroll
      for (int r = 0; r < 8; ++r) {
        int M = tM[tt] + r + (half << 3);
        int N = tN[tt] + l15;
        float bv = abias[head * n_off + bidx[M * 64 + N]];
        s_attn[M * 64 + N] = (f16)(catt[tt][r] * SCALE + bv);
      }
    }
    __syncthreads();

    // row softmax (64 rows)
    if (t < 64) {
      float mx = -1e30f;
      for (int m = 0; m < 64; ++m) mx = fmaxf(mx, (float)s_attn[t * 64 + m]);
      float sum = 0.f;
      for (int m = 0; m < 64; ++m) sum += __expf((float)s_attn[t * 64 + m] - mx);
      float inv = 1.f / sum;
      for (int m = 0; m < 64; ++m)
        s_attnh[t * 64 + m] = (f16)(__expf((float)s_attn[t * 64 + m] - mx) * inv);
    }
    __syncthreads();   // s_attn (lower 8KB of s_R) is dead past this point

#if USE_ASYNC
    // overlap: stream NEXT head's lhw slice into the dead s_lh/s_attn region
    // while this head's V x Attn^T WMMAs run. Visible after loop-top wait+barrier.
    if (head < HEADS - 1) {
#pragma unroll
      for (int r = 0; r < 2; ++r) {
        int e = (r * 256 + t) * 4;
        __builtin_amdgcn_global_load_async_to_lds_b128(
            (gv4ip)(lhw + sbase + 2048 + e), (lv4ip)(s_lh + e), 0, 0);
      }
    }
#endif

    // ll_i = V(32x64) x Attn^T(64x64): 1 tile/wave, 2 k-steps
    {
      int mi = wv >> 2, ni = wv & 3;
      int mrow = mi * 16 + l15, n = ni * 16 + l15;
      v8f dacc = {};
#pragma unroll
      for (int ks = 0; ks < 2; ++ks) {
        int kb = ks * 32;
        v16h a, b;
#pragma unroll
        for (int u = 0; u < 8; ++u) {
          a[u]     = (f16)s_lli[mrow * 64 + kb + half * 8 + u];
          a[8 + u] = (f16)s_lli[mrow * 64 + kb + 16 + half * 8 + u];
        }
        v8h blo = *(const v8h*)&s_attnh[n * 64 + kb + half * 8];
        v8h bhi = *(const v8h*)&s_attnh[n * 64 + kb + 16 + half * 8];
#pragma unroll
        for (int u = 0; u < 8; ++u) { b[u] = blo[u]; b[8 + u] = bhi[u]; }
        dacc = __builtin_amdgcn_wmma_f32_16x16x32_f16(false, a, false, b,
                                                      (short)0, dacc, false, false);
      }
      __syncthreads();  // all reads of s_lli/s_attnh done before overwrite
#pragma unroll
      for (int r = 0; r < 8; ++r) {
        int d = mi * 16 + r + (half << 3);
        int nn = ni * 16 + l15;
        float v = dacc[r];
        s_lli[d * 64 + nn] = v;                              // raw (cascade)
        s_xh[nn * 256 + head * 32 + d] = (f16)fmaxf(v, 0.f); // relu'd proj in
      }
    }
  }
  __syncthreads();

  // proj: Out(256x64) = W(256x256) x X(256x64); 8 tiles/wave, 8 k-steps
#pragma unroll
  for (int tt = 0; tt < 8; ++tt) {
    int tile = wv + tt * 8;
    int mi = tile >> 2, ni = tile & 3;
    int o = mi * 16 + l15, n = ni * 16 + l15;
    v8f acc = {};
    for (int ks = 0; ks < 8; ++ks) {
      int kb = ks * 32;
      v8h alo = *(const v8h*)&wp16[o * 256 + kb + half * 8];
      v8h ahi = *(const v8h*)&wp16[o * 256 + kb + 16 + half * 8];
      v8h blo = *(const v8h*)&s_xh[n * 256 + kb + half * 8];
      v8h bhi = *(const v8h*)&s_xh[n * 256 + kb + 16 + half * 8];
      v16h a, b;
#pragma unroll
      for (int u = 0; u < 8; ++u) { a[u] = alo[u]; a[8 + u] = ahi[u];
                                    b[u] = blo[u]; b[8 + u] = bhi[u]; }
      acc = __builtin_amdgcn_wmma_f32_16x16x32_f16(false, a, false, b,
                                                   (short)0, acc, false, false);
    }
#pragma unroll
    for (int r = 0; r < 8; ++r) {
      int oo = mi * 16 + r + (half << 3);
      int nn = ni * 16 + l15;
      llout[wbase + (size_t)oo * 64 + nn] = acc[r] + pbias[oo];
    }
  }
}

// --------------------------------------------------------------- IWT gather
__global__ __launch_bounds__(256) void iwt(
    const float* __restrict__ llo, const float* __restrict__ lh,
    const float* __restrict__ hl, const float* __restrict__ hh,
    const float* __restrict__ iwtf, float* __restrict__ out) {
  size_t tid = (size_t)blockIdx.x * 256 + threadIdx.x;
  int x = (int)(tid & 255);
  int y = (int)((tid >> 8) & 255);
  int c = (int)((tid >> 16) & 255);
  int b = (int)(tid >> 24);
  int i = y >> 1, j = x >> 1;
  int win = b * NWIN_PER_B + (i >> 3) * 16 + (j >> 3);
  size_t src = (size_t)win * 16384 + (size_t)c * 64 + ((i & 7) * 8 + (j & 7));
  const float* wf = iwtf + (size_t)c * 16 + (y & 1) * 2 + (x & 1);
  out[tid] = llo[src] * wf[0] + lh[src] * wf[4] + hl[src] * wf[8] + hh[src] * wf[12];
}

extern "C" void kernel_launch(void* const* d_in, const int* in_sizes, int n_in,
                              void* d_out, int out_size, void* d_ws, size_t ws_size,
                              hipStream_t stream) {
  const float* x    = (const float*)d_in[0];
  const float* wtf  = (const float*)d_in[1];
  const float* iwtf = (const float*)d_in[2];
  const float* dww  = (const float*)d_in[3];
  const float* dwb  = (const float*)d_in[4];
  const float* pw   = (const float*)d_in[5];
  const float* pb   = (const float*)d_in[6];
  const float* ab   = (const float*)d_in[7];
  const int*   bix  = (const int*)d_in[8];
  int n_off = in_sizes[7] / HEADS;

  float* ws    = (float*)d_ws;
  float* llw   = ws + OFF_LL;
  float* lhw   = ws + OFF_LH;
  float* hlw   = ws + OFF_HL;
  float* hhw   = ws + OFF_HH;
  float* llout = ws + OFF_LLOUT;
  f16*   wp16  = (f16*)((char*)d_ws + OFF_WP16_BYTES);

  cvt_w16<<<256, 256, 0, stream>>>(pw, wp16, DIMC * DIMC);
  wt_fwd<<<dim3(64, DIMC, 2), 256, 0, stream>>>(x, wtf, llw, lhw, hlw, hhw);
  win_attn<<<NWIN, 256, 0, stream>>>(llw, lhw, hlw, dww, dwb, wp16, pb, ab, bix,
                                     n_off, llout);
  iwt<<<131072, 256, 0, stream>>>(llout, lhw, hlw, hhw, iwtf, (float*)d_out);
}
